// Encoder_25726854103401
// MI455X (gfx1250) — matrix operand order repaired and checked by reading, not compiled, not run on hardware
//
#include <hip/hip_runtime.h>
#include <cstdint>
#include <cstddef>

// ---------------------------------------------------------------------------
// Transformer encoder block for gfx1250 (MI455X): bf16 WMMA GEMMs + flash attn
//   B=2, T=2048, C=1024, H=16, HS=64
// Async global->LDS staging (ASYNCcnt) when the toolchain exposes it.
// ---------------------------------------------------------------------------

#define KB 2
#define KT 2048
#define KC 1024
#define KH 16
#define KHS 64
#define KC4 4096
#define KBT (KB * KT)   // 4096 rows total
#define EPS 1e-5f

#if defined(__gfx1250__) && \
    __has_builtin(__builtin_amdgcn_global_load_async_to_lds_b128) && \
    __has_builtin(__builtin_amdgcn_s_wait_asynccnt)
#define USE_ASYNC 1
#else
#define USE_ASYNC 0
#endif

typedef __attribute__((ext_vector_type(16))) __bf16 bf16x16;
typedef __attribute__((ext_vector_type(8)))  float  f32x8;

union ABf {
    bf16x16 v;
    uint4   q[2];
};

static __device__ inline f32x8 f32x8_zero() {
    f32x8 z = {0.f, 0.f, 0.f, 0.f, 0.f, 0.f, 0.f, 0.f};
    return z;
}

// round-to-nearest-even f32 -> bf16 (stored as u16)
static __device__ inline unsigned short f2bf(float f) {
    unsigned int u = __float_as_uint(f);
    u += 0x7FFFu + ((u >> 16) & 1u);
    return (unsigned short)(u >> 16);
}

#if USE_ASYNC
// The builtin takes pointers to 16-byte int vectors (per hipcc diagnostic).
typedef int v4i_t __attribute__((vector_size(4 * sizeof(int))));
// Generic pointer -> address-space-typed pointers. For LDS, low 32 bits of the
// generic address are the LDS byte offset (ISA 10.2 aperture rules).
#define AS1P(p) ((__attribute__((address_space(1))) v4i_t*)(uintptr_t)(const void*)(p))
#define AS3P(p) ((__attribute__((address_space(3))) v4i_t*)(unsigned int)(uintptr_t)(const void*)(p))
#endif

// ---------------------------------------------------------------------------
// Cooperative 2048-element bf16 tile stage (128 threads, 16B granules).
// LOGW = log2(tile row width in elements): 5 -> 64x32 tile, 6 -> 32x64 tile.
// Sync variant: global_load_b128 + ds_store_b128.
// Async variant: global_load_async_to_lds_b128 (2 ops/thread, ASYNCcnt).
// ---------------------------------------------------------------------------
template <int LOGW>
static __device__ inline void stage2048_sync(unsigned short* __restrict__ dst,
                                             const unsigned short* __restrict__ src,
                                             int ld) {
    const int tid = threadIdx.x;
#pragma unroll
    for (int j = 0; j < 2; ++j) {
        const int idx = tid * 8 + j * 1024;
        const int r = idx >> LOGW;
        const int c = idx & ((1 << LOGW) - 1);
        *(uint4*)&dst[idx] = *(const uint4*)&src[(size_t)r * ld + c];
    }
}

#if USE_ASYNC
template <int LOGW>
static __device__ inline void stage2048_async(unsigned short* __restrict__ dst,
                                              const unsigned short* __restrict__ src,
                                              int ld) {
    const int tid = threadIdx.x;
#pragma unroll
    for (int j = 0; j < 2; ++j) {
        const int idx = tid * 8 + j * 1024;
        const int r = idx >> LOGW;
        const int c = idx & ((1 << LOGW) - 1);
        __builtin_amdgcn_global_load_async_to_lds_b128(
            AS1P(&src[(size_t)r * ld + c]), AS3P(&dst[idx]), 0, 0);
    }
}
#endif

// ---------------------------------------------------------------------------
// 64x64 output block MMA core: 4 waves, wave w owns rows [16w,16w+16).
// A: (64 x K) rows at caller-offset base, row stride lda.
// B: (K x 64) cols at caller-offset base, row stride ldb.
// As/Bs are double buffers: 2 x 64*32 and 2 x 32*64 bf16.
// acc[nt] = 16x16 f32 C/D fragment for columns [16*nt, 16*nt+16).
// ---------------------------------------------------------------------------
static __device__ inline void mma64x64(const unsigned short* __restrict__ A, int lda,
                                       const unsigned short* __restrict__ Bm, int ldb,
                                       int K,
                                       unsigned short* __restrict__ As,   // 2*64*32
                                       unsigned short* __restrict__ Bs,   // 2*32*64
                                       f32x8 acc[4]) {
    const int lane = threadIdx.x & 31;
    const int w    = threadIdx.x >> 5;
    const int col0 = lane & 15;
    const int hi   = lane >> 4;

#if USE_ASYNC
    // prologue: stage tile 0 into buffer 0
    stage2048_async<5>(As, A, lda);
    stage2048_async<6>(Bs, Bm, ldb);
    for (int k0 = 0; k0 < K; k0 += 32) {
        const int cur = (k0 >> 5) & 1;
        const int nxt = cur ^ 1;
        const bool more = (k0 + 32) < K;
        if (more) {  // issue next tile while current is in flight / consumed
            stage2048_async<5>(As + nxt * 2048, A + k0 + 32, lda);
            stage2048_async<6>(Bs + nxt * 2048, Bm + (size_t)(k0 + 32) * ldb, ldb);
            __builtin_amdgcn_s_wait_asynccnt(4);  // tile k0's 4 ops done (in-order)
        } else {
            __builtin_amdgcn_s_wait_asynccnt(0);
        }
        __syncthreads();  // publish tile k0 across waves
        ABf a;
        const unsigned short* Ac = As + cur * 2048;
        const unsigned short* Bc = Bs + cur * 2048;
        const int ab = (w * 16 + col0) * 32 + hi * 8;
        a.q[0] = *(const uint4*)&Ac[ab];
        a.q[1] = *(const uint4*)&Ac[ab + 16];
#pragma unroll
        for (int nt = 0; nt < 4; ++nt) {
            ABf bfr;
#pragma unroll
            for (int s = 0; s < 16; ++s)
                ((unsigned short*)&bfr)[s] = Bc[(hi * 8 + (s & 7) + (s >> 3) * 16) * 64 + nt * 16 + col0];
            acc[nt] = __builtin_amdgcn_wmma_f32_16x16x32_bf16(
                false, a.v, false, bfr.v, (short)0, acc[nt], false, false);
        }
        __syncthreads();  // all reads done before this buffer is refilled
    }
#else
    for (int k0 = 0; k0 < K; k0 += 32) {
        __syncthreads();
        stage2048_sync<5>(As, A + k0, lda);
        stage2048_sync<6>(Bs, Bm + (size_t)k0 * ldb, ldb);
        __syncthreads();
        ABf a;
        const int ab = (w * 16 + col0) * 32 + hi * 8;
        a.q[0] = *(const uint4*)&As[ab];
        a.q[1] = *(const uint4*)&As[ab + 16];
#pragma unroll
        for (int nt = 0; nt < 4; ++nt) {
            ABf bfr;
#pragma unroll
            for (int s = 0; s < 16; ++s)
                ((unsigned short*)&bfr)[s] = Bs[(hi * 8 + (s & 7) + (s >> 3) * 16) * 64 + nt * 16 + col0];
            acc[nt] = __builtin_amdgcn_wmma_f32_16x16x32_bf16(
                false, a.v, false, bfr.v, (short)0, acc[nt], false, false);
        }
    }
#endif
}

// ---------------------------------------------------------------------------
// f32 -> bf16 elementwise convert
// ---------------------------------------------------------------------------
__global__ __launch_bounds__(256) void cvt_kernel(const float* __restrict__ in,
                                                  unsigned short* __restrict__ out,
                                                  int n) {
    const int i = blockIdx.x * 256 + threadIdx.x;
    if (i < n) out[i] = f2bf(in[i]);
}

// ---------------------------------------------------------------------------
// LayerNorm over axis=1 (the SEQUENCE axis): per (b, c) column over T.
// Unbiased variance (ddof=1); eps OUTSIDE sqrt. Output bf16 [B,T,C].
// grid (C/32, B), block (32, 8)
// ---------------------------------------------------------------------------
__global__ __launch_bounds__(256) void ln_kernel(const float* __restrict__ x,
                                                 const float* __restrict__ gamma,
                                                 const float* __restrict__ beta,
                                                 unsigned short* __restrict__ out) {
    __shared__ float sSum[8][32];
    __shared__ float sSq[8][32];
    const int tx = threadIdx.x, ty = threadIdx.y;
    const int c = blockIdx.x * 32 + tx;
    const int b = blockIdx.y;
    const float* xp = x + (size_t)b * KT * KC + c;
    float s = 0.f, q = 0.f;
    for (int t = ty; t < KT; t += 8) {
        const float v = xp[(size_t)t * KC];
        s += v;
        q += v * v;
    }
    sSum[ty][tx] = s;
    sSq[ty][tx]  = q;
    __syncthreads();
    float ts = 0.f, tq = 0.f;
#pragma unroll
    for (int i = 0; i < 8; ++i) { ts += sSum[i][tx]; tq += sSq[i][tx]; }
    const float mean = ts / (float)KT;
    const float var  = (tq - (float)KT * mean * mean) / (float)(KT - 1);
    const float inv  = 1.f / (sqrtf(var) + EPS);
    const float g = gamma[c], be = beta[c];
    unsigned short* op = out + (size_t)b * KT * KC + c;
    for (int t = ty; t < KT; t += 8)
        op[(size_t)t * KC] = f2bf(g * ((xp[(size_t)t * KC] - mean) * inv) + be);
}

// ---------------------------------------------------------------------------
// QKV projections. grid (T/64, 3, B*H), block 128.
//   y=0: q = (h@Wq + bq) * HS^-0.5  -> qbf [B,H,T,HS] row-major, bf16
//   y=1: k =  h@Wk + bk            -> kT  [B,H,HS,T] (transposed), bf16
//   y=2: v =  h@Wv + bv            -> vbf [B,H,T,HS] row-major, bf16
// ---------------------------------------------------------------------------
__global__ __launch_bounds__(128) void qkv_kernel(const unsigned short* __restrict__ hbf,
                                                  const unsigned short* __restrict__ wqkv,
                                                  const float* __restrict__ bq,
                                                  const float* __restrict__ bk,
                                                  const float* __restrict__ bv,
                                                  unsigned short* __restrict__ qbf,
                                                  unsigned short* __restrict__ kT,
                                                  unsigned short* __restrict__ vbf) {
    __shared__ unsigned short As[2 * 64 * 32];
    __shared__ unsigned short Bs[2 * 32 * 64];
    const int m0    = blockIdx.x * 64;
    const int which = blockIdx.y;        // 0=q 1=k 2=v
    const int bh    = blockIdx.z;
    const int b = bh >> 4, h = bh & 15;
    const unsigned short* A  = hbf + (size_t)b * KT * KC + (size_t)m0 * KC;
    const unsigned short* Bm = wqkv + ((size_t)(which * KH + h)) * KC * KHS;

    f32x8 acc[4] = {f32x8_zero(), f32x8_zero(), f32x8_zero(), f32x8_zero()};
    mma64x64(A, KC, Bm, KHS, KC, As, Bs, acc);

    const int lane = threadIdx.x & 31, w = threadIdx.x >> 5;
    const int col0 = lane & 15, hi = lane >> 4;
    const float* bias = (which == 0) ? bq : (which == 1) ? bk : bv;
#pragma unroll
    for (int nt = 0; nt < 4; ++nt) {
#pragma unroll
        for (int r = 0; r < 8; ++r) {
            const int row = m0 + w * 16 + r + hi * 8;  // t index
            const int col = nt * 16 + col0;            // d index
            const float v = acc[nt][r] + bias[h * KHS + col];
            if (which == 0) {
                qbf[((size_t)bh * KT + row) * KHS + col] = f2bf(v * 0.125f);
            } else if (which == 1) {
                kT[((size_t)bh * KHS + col) * KT + row] = f2bf(v);
            } else {
                vbf[((size_t)bh * KT + row) * KHS + col] = f2bf(v);
            }
        }
    }
}

// ---------------------------------------------------------------------------
// Flash attention: grid (T/64, B*H), block 128 (4 waves, wave = one 16-row q tile).
// Streams s in chunks of 32; K/V chunks staged (async) in LDS.
// attn output written head-concatenated: attn[b, t, h*HS + d], bf16.
// ---------------------------------------------------------------------------
__global__ __launch_bounds__(128) void attn_kernel(const unsigned short* __restrict__ qbf,
                                                   const unsigned short* __restrict__ kT,
                                                   const unsigned short* __restrict__ vbf,
                                                   unsigned short* __restrict__ attn) {
    __shared__ unsigned short Ks[64 * 32];   // [d 0..63][s 0..31]
    __shared__ unsigned short Vs[32 * 64];   // [s 0..31][d 0..63]
    __shared__ unsigned short Pb[4][16 * 32];
    const int bh = blockIdx.y;
    const int b = bh >> 4, h = bh & 15;
    const int lane = threadIdx.x & 31, w = threadIdx.x >> 5;
    const int col0 = lane & 15, hi = lane >> 4;
    const int t0 = blockIdx.x * 64 + w * 16;

    // q A-fragments (16x64 = two K=32 chunks), pre-scaled at QKV time
    ABf qa[2];
    const unsigned short* qrow = qbf + ((size_t)bh * KT + t0 + col0) * KHS;
#pragma unroll
    for (int dc = 0; dc < 2; ++dc) {
        qa[dc].q[0] = *(const uint4*)&qrow[dc * 32 + hi * 8];
        qa[dc].q[1] = *(const uint4*)&qrow[dc * 32 + hi * 8 + 16];
    }

    f32x8 acc[4] = {f32x8_zero(), f32x8_zero(), f32x8_zero(), f32x8_zero()};
    float mrow[8], lrow[8];
#pragma unroll
    for (int r = 0; r < 8; ++r) { mrow[r] = -INFINITY; lrow[r] = 0.f; }

    for (int s0 = 0; s0 < KT; s0 += 32) {
        __syncthreads();   // previous tile's LDS reads complete
#if USE_ASYNC
        stage2048_async<5>(Ks, kT + (size_t)bh * KHS * KT + s0, KT);
        stage2048_async<6>(Vs, vbf + ((size_t)bh * KT + s0) * KHS, KHS);
        __builtin_amdgcn_s_wait_asynccnt(0);
#else
        stage2048_sync<5>(Ks, kT + (size_t)bh * KHS * KT + s0, KT);
        stage2048_sync<6>(Vs, vbf + ((size_t)bh * KT + s0) * KHS, KHS);
#endif
        __syncthreads();

        // scores: S[st] = q (16x64) @ k^T (64x16), st = s sub-tile
        f32x8 S[2];
#pragma unroll
        for (int st = 0; st < 2; ++st) {
            S[st] = f32x8_zero();
#pragma unroll
            for (int dc = 0; dc < 2; ++dc) {
                ABf kb;
#pragma unroll
                for (int s = 0; s < 16; ++s)
                    ((unsigned short*)&kb)[s] = Ks[(dc * 32 + hi * 8 + (s & 7) + (s >> 3) * 16) * 32 + st * 16 + col0];
                S[st] = __builtin_amdgcn_wmma_f32_16x16x32_bf16(
                    false, qa[dc].v, false, kb.v, (short)0, S[st], false, false);
            }
        }

        // online softmax update; P written to LDS in row-major 16x32 bf16
#pragma unroll
        for (int r = 0; r < 8; ++r) {
            float mx = fmaxf(S[0][r], S[1][r]);
#pragma unroll
            for (int m = 8; m >= 1; m >>= 1) mx = fmaxf(mx, __shfl_xor(mx, m, 32));
            const float mnew  = fmaxf(mrow[r], mx);
            const float alpha = __expf(mrow[r] - mnew);
            const float p0 = __expf(S[0][r] - mnew);
            const float p1 = __expf(S[1][r] - mnew);
            float rs = p0 + p1;
#pragma unroll
            for (int m = 8; m >= 1; m >>= 1) rs += __shfl_xor(rs, m, 32);
            lrow[r] = lrow[r] * alpha + rs;
            mrow[r] = mnew;
#pragma unroll
            for (int nt = 0; nt < 4; ++nt) acc[nt][r] *= alpha;
            Pb[w][(r + hi * 8) * 32 + col0]      = f2bf(p0);
            Pb[w][(r + hi * 8) * 32 + 16 + col0] = f2bf(p1);
        }
        __syncthreads();  // order Pb cross-lane stores before fragment reads

        // P A-fragment (16x32) and P @ V accumulate
        ABf pa;
        const int pb = col0 * 32 + hi * 8;
        pa.q[0] = *(const uint4*)&Pb[w][pb];
        pa.q[1] = *(const uint4*)&Pb[w][pb + 16];
#pragma unroll
        for (int nt = 0; nt < 4; ++nt) {
            ABf vb;
#pragma unroll
            for (int s = 0; s < 16; ++s)
                ((unsigned short*)&vb)[s] = Vs[(hi * 8 + (s & 7) + (s >> 3) * 16) * 64 + nt * 16 + col0];
            acc[nt] = __builtin_amdgcn_wmma_f32_16x16x32_bf16(
                false, pa.v, false, vb.v, (short)0, acc[nt], false, false);
        }
    }

    // normalize and store head-concatenated
#pragma unroll
    for (int r = 0; r < 8; ++r) {
        const float inv = 1.f / lrow[r];
        const int row = t0 + r + hi * 8;
#pragma unroll
        for (int nt = 0; nt < 4; ++nt) {
            const int c = h * KHS + nt * 16 + col0;
            attn[((size_t)b * KT + row) * KC + c] = f2bf(acc[nt][r] * inv);
        }
    }
}

// ---------------------------------------------------------------------------
// Generic WMMA GEMM: out(M x N) = act(A(M x K) @ B(K x N) + bias) [+ resid]
// grid (M/64, N/64), block 128. Optional ReLU, f32 and/or bf16 output.
// ---------------------------------------------------------------------------
__global__ __launch_bounds__(128) void gemm_kernel(const unsigned short* __restrict__ A,
                                                   const unsigned short* __restrict__ Bm,
                                                   const float* __restrict__ bias,
                                                   const float* __restrict__ resid,
                                                   float* __restrict__ outF,
                                                   unsigned short* __restrict__ outB,
                                                   int N, int K, int relu) {
    __shared__ unsigned short As[2 * 64 * 32];
    __shared__ unsigned short Bs[2 * 32 * 64];
    const int m0 = blockIdx.x * 64;
    const int n0 = blockIdx.y * 64;

    f32x8 acc[4] = {f32x8_zero(), f32x8_zero(), f32x8_zero(), f32x8_zero()};
    mma64x64(A + (size_t)m0 * K, K, Bm + n0, N, K, As, Bs, acc);

    const int lane = threadIdx.x & 31, w = threadIdx.x >> 5;
    const int col0 = lane & 15, hi = lane >> 4;
#pragma unroll
    for (int nt = 0; nt < 4; ++nt) {
#pragma unroll
        for (int r = 0; r < 8; ++r) {
            const int gr = m0 + w * 16 + r + hi * 8;
            const int gc = n0 + nt * 16 + col0;
            float v = acc[nt][r];
            if (bias)  v += bias[gc];
            if (relu)  v = fmaxf(v, 0.f);
            if (resid) v += resid[(size_t)gr * N + gc];
            if (outF)  outF[(size_t)gr * N + gc] = v;
            if (outB)  outB[(size_t)gr * N + gc] = f2bf(v);
        }
    }
}

// ---------------------------------------------------------------------------
// Host-side launcher
// ---------------------------------------------------------------------------
extern "C" void kernel_launch(void* const* d_in, const int* in_sizes, int n_in,
                              void* d_out, int out_size, void* d_ws, size_t ws_size,
                              hipStream_t stream) {
    (void)in_sizes; (void)n_in; (void)out_size; (void)ws_size;
    const float* x   = (const float*)d_in[0];
    const float* Wq  = (const float*)d_in[1];
    const float* bq  = (const float*)d_in[2];
    const float* Wk  = (const float*)d_in[3];
    const float* bk  = (const float*)d_in[4];
    const float* Wv  = (const float*)d_in[5];
    const float* bv  = (const float*)d_in[6];
    const float* Wo  = (const float*)d_in[7];
    const float* bo  = (const float*)d_in[8];
    const float* W1  = (const float*)d_in[9];
    const float* b1  = (const float*)d_in[10];
    const float* W2  = (const float*)d_in[11];
    const float* b2  = (const float*)d_in[12];
    const float* g1  = (const float*)d_in[13];
    const float* be1 = (const float*)d_in[14];
    const float* g2  = (const float*)d_in[15];
    const float* be2 = (const float*)d_in[16];
    float* out = (float*)d_out;

    // workspace bump allocation (256B aligned), ~120 MB total
    char* wsp = (char*)d_ws;
    size_t off = 0;
    auto walloc = [&](size_t bytes) -> void* {
        void* p = wsp + off;
        off += (bytes + 255) & ~(size_t)255;
        return p;
    };
    const size_t HCHS = (size_t)KH * KC * KHS;       // 1M elems per qkv weight
    unsigned short* wqkv = (unsigned short*)walloc(3 * HCHS * 2);
    unsigned short* wob  = (unsigned short*)walloc((size_t)KC * KC * 2);
    unsigned short* w1b  = (unsigned short*)walloc((size_t)KC * KC4 * 2);
    unsigned short* w2b  = (unsigned short*)walloc((size_t)KC4 * KC * 2);
    unsigned short* hbf  = (unsigned short*)walloc((size_t)KB * KT * KC * 2);
    unsigned short* qbf  = (unsigned short*)walloc((size_t)KB * KH * KT * KHS * 2);
    unsigned short* kTb  = (unsigned short*)walloc((size_t)KB * KH * KT * KHS * 2);
    unsigned short* vbf  = (unsigned short*)walloc((size_t)KB * KH * KT * KHS * 2);
    unsigned short* att  = (unsigned short*)walloc((size_t)KB * KT * KC * 2);
    float*          x2f  = (float*)walloc((size_t)KB * KT * KC * 4);
    unsigned short* h2b  = (unsigned short*)walloc((size_t)KB * KT * KC * 2);
    unsigned short* ybf  = (unsigned short*)walloc((size_t)KB * KT * KC4 * 2);

    // 1) weight conversions to bf16
    cvt_kernel<<<(int)(HCHS / 256), 256, 0, stream>>>(Wq, wqkv, (int)HCHS);
    cvt_kernel<<<(int)(HCHS / 256), 256, 0, stream>>>(Wk, wqkv + HCHS, (int)HCHS);
    cvt_kernel<<<(int)(HCHS / 256), 256, 0, stream>>>(Wv, wqkv + 2 * HCHS, (int)HCHS);
    cvt_kernel<<<(KC * KC) / 256, 256, 0, stream>>>(Wo, wob, KC * KC);
    cvt_kernel<<<(KC * KC4) / 256, 256, 0, stream>>>(W1, w1b, KC * KC4);
    cvt_kernel<<<(KC4 * KC) / 256, 256, 0, stream>>>(W2, w2b, KC4 * KC);

    // 2) LN1 (over sequence axis) -> h (bf16)
    ln_kernel<<<dim3(KC / 32, KB), dim3(32, 8), 0, stream>>>(x, g1, be1, hbf);

    // 3) QKV projections (q scaled, k transposed)
    qkv_kernel<<<dim3(KT / 64, 3, KB * KH), 128, 0, stream>>>(
        hbf, wqkv, bq, bk, bv, qbf, kTb, vbf);

    // 4) flash attention -> head-concat attn (bf16)
    attn_kernel<<<dim3(KT / 64, KB * KH), 128, 0, stream>>>(qbf, kTb, vbf, att);

    // 5) output projection + residual: x2 = x + attn@Wo + bo (f32)
    gemm_kernel<<<dim3(KBT / 64, KC / 64), 128, 0, stream>>>(
        att, wob, bo, x, x2f, nullptr, KC, KC, 0);

    // 6) LN2 -> h2 (bf16)
    ln_kernel<<<dim3(KC / 32, KB), dim3(32, 8), 0, stream>>>(x2f, g2, be2, h2b);

    // 7) FFN1: y = relu(h2@W1 + b1) (bf16)
    gemm_kernel<<<dim3(KBT / 64, KC4 / 64), 128, 0, stream>>>(
        h2b, w1b, b1, nullptr, nullptr, ybf, KC4, KC, 1);

    // 8) FFN2 + residual: out = x2 + y@W2 + b2 (f32 -> d_out)
    gemm_kernel<<<dim3(KBT / 64, KC / 64), 128, 0, stream>>>(
        ybf, w2b, b2, x2f, out, nullptr, KC, KC4, 0);
}
